// FlexAttention_33380485824834
// MI455X (gfx1250) — matrix operand order repaired and checked by reading, not compiled, not run on hardware
//
#include <hip/hip_runtime.h>

// ---------------- problem constants (match reference) ----------------
#define NUM_HEADS    32
#define NUM_KV_HEADS 8
#define HEAD_DIM     128
#define SEQ          2048
#define GQA          (NUM_HEADS / NUM_KV_HEADS)   // 4
#define SOFTCAP      50.0f
#define QSCALE       0.08838834764831845f         // 1/sqrt(128)

// ---------------- tiling ----------------
#define NWAVES  4            // waves per block (wave32)
#define QTILE   (NWAVES*16)  // 64 q rows per block
#define KVTILE  32           // kv rows per iteration

typedef __attribute__((ext_vector_type(16))) _Float16 v16h;
typedef __attribute__((ext_vector_type(8)))  _Float16 v8h;
typedef __attribute__((ext_vector_type(2)))  _Float16 h2;
typedef __attribute__((ext_vector_type(8)))  float    v8f;

union V16H { v16h v; v8h h[2]; h2 p[8]; _Float16 e[16]; };
union V8H  { v8h  v; h2 p[4];  _Float16 e[8]; };

static __device__ __forceinline__ v8f wmma16(v16h a, v16h b, v8f c) {
  // D = A(16x32 f16) * B(32x16 f16) + C(16x16 f32)
  return __builtin_amdgcn_wmma_f32_16x16x32_f16(
      /*neg_a=*/false, a, /*neg_b=*/false, b,
      /*c_mod=*/(short)0, c, /*reuse_a=*/false, /*reuse_b=*/false);
}

// packed f32->f16 convert (v_cvt_pk_f16_f32, RTZ); builtin returns an
// __fp16 vector, bit_cast to our _Float16 vector type
static __device__ __forceinline__ h2 pkrtz(float a, float b) {
  return __builtin_bit_cast(h2, __builtin_amdgcn_cvt_pkrtz(a, b));
}

// cross-lane XOR shuffle with zero VALU overhead: ds_swizzle_b32, group-of-32
// mode, imm = (xor_mask<<10) | (or_mask<<5) | and_mask ; requires full EXEC.
#define SWZ_XOR(x, mask) \
  __builtin_bit_cast(float, __builtin_amdgcn_ds_swizzle( \
      __builtin_bit_cast(int, (float)(x)), ((mask) << 10) | 0x1f))

static __device__ __forceinline__ float softcap(float x) {
  // 50*tanh(x/50) = 50 - 100/(exp(x/25)+1)
  // v_mul + v_exp_f32 + v_add + v_rcp_f32 + v_fma (no IEEE-div sequence)
  const float e = __expf(x * (1.0f / 25.0f));
  return __builtin_fmaf(-2.0f * SOFTCAP, __builtin_amdgcn_rcpf(e + 1.0f), SOFTCAP);
}

__global__ __launch_bounds__(NWAVES * 32)
void fa_fwd_kernel(const float* __restrict__ Q, const float* __restrict__ K,
                   const float* __restrict__ V, float* __restrict__ Out)
{
  // double-buffered K/V tiles (f16), plus per-wave P bounce buffer
  __shared__ __align__(16) _Float16 Kbuf[2][KVTILE * HEAD_DIM];   // [kv][d]  2x8KB
  __shared__ __align__(16) _Float16 Vbuf[2][HEAD_DIM * KVTILE];   // [d][kv]  2x8KB (transposed)
  __shared__ __align__(16) _Float16 Psh[NWAVES][16 * KVTILE];     // 4KB

  const int tid    = threadIdx.x;
  const int lane   = tid & 31;
  const int wave   = tid >> 5;
  const int lanelo = lane & 15;
  const int hi     = lane >> 4;           // which 16-lane half

  const int qtile = blockIdx.x;
  const int h     = blockIdx.y;
  const int b     = blockIdx.z;
  const int kvh   = h / GQA;

  const int qbase = qtile * QTILE;
  const int qrow0 = qbase + wave * 16;    // this wave's first q row

  // ---- load Q as 4 A-fragments (16x32 f16 each), pre-scaled ----
  // A layout (16-bit, 16x32): lane L (M=L%16), half i -> K = (i/8)*16 + (L>=16)*8 + (i%8)
  const float* Qrow = Q + (((size_t)b * NUM_HEADS + h) * SEQ + (size_t)(qrow0 + lanelo)) * HEAD_DIM;
  V16H qa[4];
  #pragma unroll
  for (int c = 0; c < 4; ++c) {
    const int d0 = c * 32 + hi * 8;        // halves 0..7
    const int d1 = c * 32 + 16 + hi * 8;   // halves 8..15
    #pragma unroll
    for (int j = 0; j < 4; ++j)
      qa[c].p[j] = pkrtz(Qrow[d0 + 2*j] * QSCALE, Qrow[d0 + 2*j + 1] * QSCALE);
    #pragma unroll
    for (int j = 0; j < 4; ++j)
      qa[c].p[4 + j] = pkrtz(Qrow[d1 + 2*j] * QSCALE, Qrow[d1 + 2*j + 1] * QSCALE);
  }

  // ---- flash accumulators: rows r+8*hi live in this lane ----
  v8f   o[8];
  float m[8], l[8];
  #pragma unroll
  for (int n = 0; n < 8; ++n) o[n] = (v8f)0.0f;
  #pragma unroll
  for (int r = 0; r < 8; ++r) { m[r] = -__builtin_inff(); l[r] = 0.0f; }

  const float* Kbase = K + ((size_t)b * NUM_KV_HEADS + kvh) * SEQ * HEAD_DIM;
  const float* Vbase = V + ((size_t)b * NUM_KV_HEADS + kvh) * SEQ * HEAD_DIM;

  // ---- staging partition (128 threads) ----
  // K: thread owns 1 kv row x 32 d cols  -> 8x b128 global, 4x b128 LDS store
  // V: thread owns 8 kv rows x 4 d cols  -> 8x b128 global, 4x b128 LDS store (transposed)
  const int krow  = tid >> 2;          // 0..31
  const int kcol  = (tid & 3) * 32;    // 0/32/64/96
  const int vcol0 = (tid & 31) * 4;    // 0..124
  const int vrow0 = (tid >> 5) * 8;    // 0/8/16/24

  float kreg[32], vreg[32];

  auto stage_load = [&](int kv0) {
    const float* kp = Kbase + (size_t)(kv0 + krow) * HEAD_DIM + kcol;
    #pragma unroll
    for (int j = 0; j < 32; ++j) kreg[j] = kp[j];
    #pragma unroll
    for (int rr = 0; rr < 8; ++rr) {
      const float* vp = Vbase + (size_t)(kv0 + vrow0 + rr) * HEAD_DIM + vcol0;
      #pragma unroll
      for (int cc = 0; cc < 4; ++cc) vreg[rr * 4 + cc] = vp[cc];
    }
  };
  auto stage_store = [&](int bufi) {
    v8h* kd = (v8h*)&Kbuf[bufi][krow * HEAD_DIM + kcol];
    #pragma unroll
    for (int j = 0; j < 4; ++j) {
      V8H kk;
      #pragma unroll
      for (int j2 = 0; j2 < 4; ++j2)
        kk.p[j2] = pkrtz(kreg[j * 8 + 2*j2], kreg[j * 8 + 2*j2 + 1]);
      kd[j] = kk.v;
    }
    #pragma unroll
    for (int dd = 0; dd < 4; ++dd) {
      V8H w;   // column dd: 8 consecutive kv rows -> contiguous in [d][kv]
      #pragma unroll
      for (int rr = 0; rr < 4; ++rr)
        w.p[rr] = pkrtz(vreg[(2*rr) * 4 + dd], vreg[(2*rr + 1) * 4 + dd]);
      *(v8h*)&Vbuf[bufi][(vcol0 + dd) * KVTILE + vrow0] = w.v;
    }
  };

  const int ntiles = (qbase + QTILE) / KVTILE;   // causal upper bound, always >= 2

  stage_load(0);
  stage_store(0);
  __syncthreads();

  int cur = 0;
  for (int t = 0; t < ntiles; ++t) {
    const int kv0 = t * KVTILE;

    // prefetch next tile into registers; WMMA block below hides the latency
    if (t + 1 < ntiles) stage_load(kv0 + KVTILE);

    if (kv0 <= qrow0 + 15) {  // wave-uniform: skip tiles fully above our rows
      const _Float16* Ksh = Kbuf[cur];
      const _Float16* Vsh = Vbuf[cur];

      // ---------------- S = Qs * K^T (2 N-chunks x 4 K-chunks) ----------------
      v8f s[2];
      #pragma unroll
      for (int nn = 0; nn < 2; ++nn) {
        s[nn] = (v8f)0.0f;
        #pragma unroll
        for (int c = 0; c < 4; ++c) {
          // B layout (32x16 f16): lane L -> N=L%16, half i -> K=(i/8)*16+(L/16)*8+(i%8)
          V16H kb;
          const v8h* ks = (const v8h*)&Ksh[(nn * 16 + lanelo) * HEAD_DIM + c * 32 + hi * 8];
          kb.h[0] = ks[0];
          kb.h[1] = ks[2];
          s[nn] = wmma16(qa[c].v, kb.v, s[nn]);
        }
      }

      // ---------------- softcap + causal mask + online softmax ----------------
      float p0[8], p1[8], alpha[8];
      const int kvg0 = kv0 + lanelo;
      const int kvg1 = kv0 + 16 + lanelo;
      #pragma unroll
      for (int r = 0; r < 8; ++r) {
        const int qg = qrow0 + r + 8 * hi;
        float x0 = softcap(s[0][r]);
        float x1 = softcap(s[1][r]);
        x0 = (kvg0 <= qg) ? x0 : -__builtin_inff();
        x1 = (kvg1 <= qg) ? x1 : -__builtin_inff();
        float mx = fmaxf(x0, x1);
        mx = fmaxf(mx, SWZ_XOR(mx, 1));
        mx = fmaxf(mx, SWZ_XOR(mx, 2));
        mx = fmaxf(mx, SWZ_XOR(mx, 4));
        mx = fmaxf(mx, SWZ_XOR(mx, 8));
        const float mn = fmaxf(m[r], mx);
        const float al = (m[r] == mn) ? 1.0f : __expf(m[r] - mn);
        const float sm = (mn == -__builtin_inff()) ? 0.0f : mn;
        const float e0 = __expf(x0 - sm);   // -inf -> 0
        const float e1 = __expf(x1 - sm);
        float rs = e0 + e1;
        rs += SWZ_XOR(rs, 1);
        rs += SWZ_XOR(rs, 2);
        rs += SWZ_XOR(rs, 4);
        rs += SWZ_XOR(rs, 8);
        l[r] = l[r] * al + rs;
        m[r] = mn;
        alpha[r] = al;
        p0[r] = e0;
        p1[r] = e1;
      }

      // rescale O by alpha (row-wise)
      #pragma unroll
      for (int n = 0; n < 8; ++n)
        #pragma unroll
        for (int r = 0; r < 8; ++r) o[n][r] *= alpha[r];

      // ---------------- P: C-layout -> A-layout via wave-private LDS ----------------
      _Float16* pp = &Psh[wave][0];
      #pragma unroll
      for (int r = 0; r < 8; ++r) {
        const h2 ph = pkrtz(p0[r], p1[r]);                        // one packed cvt
        pp[(r + 8 * hi) * KVTILE + lanelo]      = ph.x;           // ds_store_b16
        pp[(r + 8 * hi) * KVTILE + 16 + lanelo] = ph.y;           // ds_store_b16_d16_hi
      }
      asm volatile("s_wait_dscnt 0" ::: "memory");   // wave-internal RAW on LDS
      V16H pa;
      pa.h[0] = *(const v8h*)&Psh[wave][lanelo * KVTILE + hi * 8];        // kv = hi*8..+7
      pa.h[1] = *(const v8h*)&Psh[wave][lanelo * KVTILE + 16 + hi * 8];   // kv = 16+hi*8..+7

      // ---------------- O += P * V (8 N-chunks over d) ----------------
      #pragma unroll
      for (int n = 0; n < 8; ++n) {
        V16H vb;   // B[kv][d] from transposed Vbuf: contiguous in kv
        const v8h* vs = (const v8h*)&Vsh[(n * 16 + lanelo) * KVTILE + hi * 8];
        vb.h[0] = vs[0];
        vb.h[1] = vs[2];
        o[n] = wmma16(pa.v, vb.v, o[n]);
      }
    }

    // convert + publish next tile to the alternate buffer
    if (t + 1 < ntiles) stage_store(cur ^ 1);
    __syncthreads();
    cur ^= 1;
  }

  // ---------------- epilogue: normalize and store [B, q, H*d] fp32 ----------------
  #pragma unroll
  for (int r = 0; r < 8; ++r) {
    const int q = qrow0 + r + 8 * hi;
    const float inv = __builtin_amdgcn_rcpf(l[r]);   // v_rcp_f32, no IEEE-div sequence
    float* op = Out + ((size_t)b * SEQ + q) * (NUM_HEADS * HEAD_DIM) + h * HEAD_DIM;
    #pragma unroll
    for (int n = 0; n < 8; ++n)
      op[n * 16 + lanelo] = o[n][r] * inv;
  }
}

extern "C" void kernel_launch(void* const* d_in, const int* in_sizes, int n_in,
                              void* d_out, int out_size, void* d_ws, size_t ws_size,
                              hipStream_t stream) {
  const float* Q = (const float*)d_in[0];
  const float* K = (const float*)d_in[1];
  const float* V = (const float*)d_in[2];
  // d_in[3] is the additive causal mask; causality is applied analytically.
  float* Out = (float*)d_out;

  const int B = in_sizes[0] / (NUM_HEADS * SEQ * HEAD_DIM);
  dim3 grid(SEQ / QTILE, NUM_HEADS, B);
  dim3 block(NWAVES * 32);
  fa_fwd_kernel<<<grid, block, 0, stream>>>(Q, K, V, Out);
}
